// FilterbankSubEdge_30494267802068
// MI455X (gfx1250) — matrix operand order repaired and checked by reading, not compiled, not run on hardware
//
#include <hip/hip_runtime.h>
#include <hip/hip_bf16.h>

// Problem constants (from the reference)
#define B_SZ   128
#define C_SZ   21
#define L_SZ   2048
#define W_SZ   32
#define S_SZ   8
#define O_SZ   16
#define NW_SZ  252
#define NK_SZ  (NW_SZ * O_SZ)        // 4032
#define K_SZ   (C_SZ * W_SZ)         // 672
#define KSTEPS (K_SZ / 32)           // 21 (one channel c per WMMA K-step)
#define NX_EL  (B_SZ * C_SZ * L_SZ)  // 5,505,024 x elements
#define NWT_EL (NK_SZ * K_SZ)        // 2,709,504 weight elements

typedef __attribute__((ext_vector_type(16))) __bf16 v16bf;
typedef __attribute__((ext_vector_type(8)))  __bf16 bf16x8;
typedef __attribute__((ext_vector_type(8)))  float  v8f;
typedef __attribute__((ext_vector_type(4)))  float  f32x4;

// ---------------------------------------------------------------- helpers ---
static __device__ __forceinline__ __bf16 f32_to_bf16_rne(float f) {
  unsigned u = __builtin_bit_cast(unsigned, f);
  u += 0x7FFFu + ((u >> 16) & 1u);          // round-to-nearest-even
  return __builtin_bit_cast(__bf16, (unsigned short)(u >> 16));
}
static __device__ __forceinline__ float bf16_to_f32(__bf16 b) {
  unsigned u = ((unsigned)__builtin_bit_cast(unsigned short, b)) << 16;
  return __builtin_bit_cast(float, u);
}
static __device__ __forceinline__ void split1(float f, __bf16& h, __bf16& l) {
  __bf16 hb = f32_to_bf16_rne(f);
  h = hb;
  l = f32_to_bf16_rne(f - bf16_to_f32(hb));
}

static __device__ __forceinline__ v8f wmma_bf16(v16bf a, v16bf b, v8f c) {
  return __builtin_amdgcn_wmma_f32_16x16x32_bf16(
      /*neg_a=*/false, a, /*neg_b=*/false, b,
      /*c_mod=*/(short)0, c, /*reuse_a=*/false, /*reuse_b=*/false);
}

// Concatenate two 8-element bf16 chunks (two b128 loads) into a WMMA fragment.
static __device__ __forceinline__ v16bf ld16(const __bf16* p0, const __bf16* p1) {
  bf16x8 a = *(const bf16x8*)p0;
  bf16x8 b = *(const bf16x8*)p1;
  return __builtin_shufflevector(a, b, 0, 1, 2, 3, 4, 5, 6, 7,
                                 8, 9, 10, 11, 12, 13, 14, 15);
}

// ------------------------------------------------- pass 1: bf16 hi/lo split ---
// Memory-bound one-time conversion: each fp32 element split exactly once.
// 8 elements per thread, 16B stores.
__global__ __launch_bounds__(256)
void bf16_split_pass(const float* __restrict__ src,
                     __bf16* __restrict__ hi, __bf16* __restrict__ lo, int n8) {
  int i = blockIdx.x * blockDim.x + threadIdx.x;
  if (i >= n8) return;
  const f32x4* s = (const f32x4*)src + (size_t)i * 2;
  f32x4 v0 = s[0];
  f32x4 v1 = s[1];
  bf16x8 h, l;
#pragma unroll
  for (int k = 0; k < 4; ++k) {
    __bf16 hh, ll;
    split1(v0[k], hh, ll); h[k] = hh;     l[k] = ll;
    split1(v1[k], hh, ll); h[4 + k] = hh; l[4 + k] = ll;
  }
  ((bf16x8*)hi)[i] = h;
  ((bf16x8*)lo)[i] = l;
}

// ------------------------------------------- pass 2: pure load+WMMA GEMMs ---
// One block per window j; 4 waves; wave w owns batch rows [32w, 32w+32) as two
// 16x16 output tiles sharing one weight fragment pair per K-step.
// Inner loop: 12x global_load_b128 + 6x v_wmma, no conversion VALU.
__global__ __launch_bounds__(128)
void fbank_wmma_main(const __bf16* __restrict__ xhi, const __bf16* __restrict__ xlo,
                     const __bf16* __restrict__ whi, const __bf16* __restrict__ wlo,
                     float* __restrict__ out) {
  const int j    = blockIdx.x;
  const int lane = threadIdx.x & 31;
  const int wave = threadIdx.x >> 5;
  const int half = lane >> 4;
  const int mn   = lane & 15;
  const int s    = (j == NW_SZ - 1) ? (L_SZ - W_SZ) : (j * S_SZ);

  // A-fragment: half h reads w in {16h..16h+15} (contiguous, matches B).
  size_t ao0 = (size_t)(wave * 32 + mn) * (C_SZ * L_SZ) + s + half * 16;
  size_t ao1 = ao0 + (size_t)16 * (C_SZ * L_SZ);
  // B-fragment: lanes 16-31 take K=16..31 of the step (contiguous 16 elements).
  size_t bo  = (size_t)(j * O_SZ + mn) * K_SZ + half * 16;

  v8f acc0 = {}; v8f acc1 = {};

  for (int ks = 0; ks < KSTEPS; ++ks) {
    v16bf bhi  = ld16(whi + bo,      whi + bo + 8);
    v16bf blo  = ld16(wlo + bo,      wlo + bo + 8);
    v16bf ahi0 = ld16(xhi + ao0,     xhi + ao0 + 8);
    v16bf alo0 = ld16(xlo + ao0,     xlo + ao0 + 8);
    v16bf ahi1 = ld16(xhi + ao1,     xhi + ao1 + 8);
    v16bf alo1 = ld16(xlo + ao1,     xlo + ao1 + 8);

    if (ks + 1 < KSTEPS) {                 // next channel row is +4KB away
      __builtin_prefetch(xhi + ao0 + L_SZ, 0, 3);
      __builtin_prefetch(xhi + ao1 + L_SZ, 0, 3);
      __builtin_prefetch(xlo + ao0 + L_SZ, 0, 3);
      __builtin_prefetch(xlo + ao1 + L_SZ, 0, 3);
    }

    // bf16x3: hi*hi + lo*hi + hi*lo, f32 accumulate; interleave the two
    // independent accumulators to keep the matrix pipe busy.
    acc0 = wmma_bf16(ahi0, bhi, acc0);
    acc1 = wmma_bf16(ahi1, bhi, acc1);
    acc0 = wmma_bf16(alo0, bhi, acc0);
    acc1 = wmma_bf16(alo1, bhi, acc1);
    acc0 = wmma_bf16(ahi0, blo, acc0);
    acc1 = wmma_bf16(ahi1, blo, acc1);

    ao0 += L_SZ; ao1 += L_SZ; bo += 32;
  }

  // C/D layout: VGPR r holds M=r (lanes 0-15) / M=r+8 (lanes 16-31); N=lane%16.
  const int col    = j * O_SZ + mn;
  const int rbase0 = wave * 32 + half * 8;
#pragma unroll
  for (int r = 0; r < 8; ++r) {
    out[(size_t)(rbase0 + r)      * NK_SZ + col] = acc0[r];
    out[(size_t)(rbase0 + 16 + r) * NK_SZ + col] = acc1[r];
  }
}

// --------------------------- fallback: fused single-pass (if ws too small) ---
static __device__ __forceinline__ void split16(const f32x4 q0, const f32x4 q1,
                                               const f32x4 q2, const f32x4 q3,
                                               v16bf& hi, v16bf& lo) {
  float f[16];
#pragma unroll
  for (int i = 0; i < 4; ++i) { f[i] = q0[i]; f[4+i] = q1[i]; f[8+i] = q2[i]; f[12+i] = q3[i]; }
#pragma unroll
  for (int i = 0; i < 16; ++i) {
    __bf16 h, l;
    split1(f[i], h, l);
    hi[i] = h; lo[i] = l;
  }
}

__global__ __launch_bounds__(128)
void fbank_wmma_fused(const float* __restrict__ x, const float* __restrict__ wgt,
                      float* __restrict__ out) {
  const int j    = blockIdx.x;
  const int lane = threadIdx.x & 31;
  const int wave = threadIdx.x >> 5;
  const int half = lane >> 4;
  const int mn   = lane & 15;
  const int s    = (j == NW_SZ - 1) ? (L_SZ - W_SZ) : (j * S_SZ);

  const float* bp  = wgt + (size_t)(j * O_SZ + mn) * K_SZ + half * 16;
  const float* ap0 = x + (size_t)(wave * 32 + mn) * (C_SZ * L_SZ) + s + half * 16;
  const float* ap1 = ap0 + (size_t)16 * (C_SZ * L_SZ);

  v8f acc0 = {}; v8f acc1 = {};

  for (int ks = 0; ks < KSTEPS; ++ks) {
    const float* b  = bp  + ks * 32;
    const float* a0 = ap0 + ks * L_SZ;
    const float* a1 = ap1 + ks * L_SZ;

    v16bf bhi, blo, ahi0, alo0, ahi1, alo1;
    split16(*(const f32x4*)(b),      *(const f32x4*)(b + 4),
            *(const f32x4*)(b + 8),  *(const f32x4*)(b + 12), bhi, blo);
    split16(*(const f32x4*)(a0),     *(const f32x4*)(a0 + 4),
            *(const f32x4*)(a0 + 8), *(const f32x4*)(a0 + 12), ahi0, alo0);
    split16(*(const f32x4*)(a1),     *(const f32x4*)(a1 + 4),
            *(const f32x4*)(a1 + 8), *(const f32x4*)(a1 + 12), ahi1, alo1);

    acc0 = wmma_bf16(ahi0, bhi, acc0);
    acc1 = wmma_bf16(ahi1, bhi, acc1);
    acc0 = wmma_bf16(alo0, bhi, acc0);
    acc1 = wmma_bf16(alo1, bhi, acc1);
    acc0 = wmma_bf16(ahi0, blo, acc0);
    acc1 = wmma_bf16(ahi1, blo, acc1);
  }

  const int col    = j * O_SZ + mn;
  const int rbase0 = wave * 32 + half * 8;
#pragma unroll
  for (int r = 0; r < 8; ++r) {
    out[(size_t)(rbase0 + r)      * NK_SZ + col] = acc0[r];
    out[(size_t)(rbase0 + 16 + r) * NK_SZ + col] = acc1[r];
  }
}

// ------------------------------------------------------------------ launch ---
extern "C" void kernel_launch(void* const* d_in, const int* in_sizes, int n_in,
                              void* d_out, int out_size, void* d_ws, size_t ws_size,
                              hipStream_t stream) {
  (void)in_sizes; (void)n_in; (void)out_size;
  const float* x   = (const float*)d_in[0];   // (B, C, L) fp32
  const float* wgt = (const float*)d_in[1];   // (NK, W*C) fp32
  float*       out = (float*)d_out;           // (B, NK) fp32

  const size_t need = (size_t)2 * (NX_EL + NWT_EL) * sizeof(unsigned short);

  if (ws_size >= need && d_ws != nullptr) {
    __bf16* xhi = (__bf16*)d_ws;
    __bf16* xlo = xhi + NX_EL;
    __bf16* whi = xlo + NX_EL;
    __bf16* wlo = whi + NWT_EL;

    const int nx8 = NX_EL / 8, nw8 = NWT_EL / 8;
    hipLaunchKernelGGL(bf16_split_pass, dim3((nx8 + 255) / 256), dim3(256), 0,
                       stream, x, xhi, xlo, nx8);
    hipLaunchKernelGGL(bf16_split_pass, dim3((nw8 + 255) / 256), dim3(256), 0,
                       stream, wgt, whi, wlo, nw8);
    hipLaunchKernelGGL(fbank_wmma_main, dim3(NW_SZ), dim3(128), 0, stream,
                       xhi, xlo, whi, wlo, out);
  } else {
    hipLaunchKernelGGL(fbank_wmma_fused, dim3(NW_SZ), dim3(128), 0, stream,
                       x, wgt, out);
  }
}
